// Vit_Channel_Map_1992864825605
// MI455X (gfx1250) — matrix operand order repaired and checked by reading, not compiled, not run on hardware
//
#include <hip/hip_runtime.h>
#include <hip/hip_bf16.h>
#include <math.h>

// ---------------------------------------------------------------------------
// Problem constants: S=128, NF=64, B=16, N=1 -> rows M = 16*64 = 1024
//   GEMM1: xe     = x  @ We^T   M=1024 N=1024  K=16384   (+be +pos)
//   GEMM2: scaled = xe @ Wv^T   M=1024 N=1024  K=1024    (* k gate)
//   GEMM3: out    = sc @ Wo^T   M=1024 N=16384 K=1024    (+bo)
// All TN (K contiguous in both operands) -> ideal for v_wmma_f32_16x16x32_bf16
// ---------------------------------------------------------------------------

typedef __attribute__((ext_vector_type(16))) __bf16 v16bf;
typedef __attribute__((ext_vector_type(8)))  float  v8f;

union FragBF { v16bf v; uint4 q[2]; };

// fp32 pair -> packed bf16 dword in a single v_perm_b32 (truncation).
// Byte pool: {hi(bytes 4-7), lo(bytes 0-3)}; select lo[3:2] then hi[3:2].
__device__ __forceinline__ unsigned pack_bf16_pair(float lo, float hi) {
  return __builtin_amdgcn_perm(__float_as_uint(hi), __float_as_uint(lo), 0x07060302u);
}

#define BM 128
#define BN 128
#define BK 32
#define LDS_PITCH   80                    // 32 bf16 = 64B + 16B pad (conflict spread, 16B aligned)
#define TILE_BYTES  (128 * LDS_PITCH)     // 10240
#define STAGE_BYTES (2 * TILE_BYTES)      // A tile + B tile = 20480

// MODE 0: C = A@B^T + bias[n] + pos[(m&63)*1024+n]          (xe)
// MODE 1: C = (A@B^T) * kk[(m>>6)*512 + (n>>7)*64 + (m&63)] (k-gated v)
// MODE 2: C = A@B^T + bias[n]                               (final out)
template<int MODE>
__launch_bounds__(256)
__global__ void gemm_tn_wmma_bf16(const float* __restrict__ A,
                                  const float* __restrict__ B,
                                  int K, int N,
                                  float* __restrict__ Cout,
                                  const float* __restrict__ bias,
                                  const float* __restrict__ pos,
                                  const float* __restrict__ kk)
{
  __shared__ __align__(16) char lds[2 * STAGE_BYTES];   // double buffered: 40 KB

  const int tid   = threadIdx.x;
  const int wid   = tid >> 5;          // 8 waves
  const int lane  = tid & 31;
  const int wm    = wid & 3;           // 4 waves along M: 32 rows each
  const int wn    = wid >> 2;          // 2 waves along N: 64 cols each
  const int lmod  = lane & 15;
  const int lhalf = lane >> 4;

  const int bm = blockIdx.y, bn = blockIdx.x;
  const float* Abase = A + (size_t)bm * BM * K;
  const float* Bbase = B + (size_t)bn * BN * K;

  // 256 threads move a 128x32 fp32 tile; each thread owns 2 chunks of 8
  // consecutive floats per tile -> 4 float4 global loads, 2 ds_store_b128.
  float4 ra[4], rb[4];
  int rrow[2], rc8[2];
  #pragma unroll
  for (int i = 0; i < 2; ++i) { int ci = tid + i * 256; rrow[i] = ci >> 2; rc8[i] = ci & 3; }

  auto gload = [&](int kt) {
    const float* Ak = Abase + (size_t)kt * BK;
    const float* Bk = Bbase + (size_t)kt * BK;
    #pragma unroll
    for (int i = 0; i < 2; ++i) {
      const size_t off = (size_t)rrow[i] * K + rc8[i] * 8;
      ra[2*i+0] = *(const float4*)(Ak + off);
      ra[2*i+1] = *(const float4*)(Ak + off + 4);
      rb[2*i+0] = *(const float4*)(Bk + off);
      rb[2*i+1] = *(const float4*)(Bk + off + 4);
    }
  };
  auto lstore = [&](int s) {   // fp32 -> bf16 convert-on-store into LDS
    char* la = lds + s * STAGE_BYTES;
    char* lb = la + TILE_BYTES;
    #pragma unroll
    for (int i = 0; i < 2; ++i) {
      uint4 ua, ub;
      ua.x = pack_bf16_pair(ra[2*i+0].x, ra[2*i+0].y);
      ua.y = pack_bf16_pair(ra[2*i+0].z, ra[2*i+0].w);
      ua.z = pack_bf16_pair(ra[2*i+1].x, ra[2*i+1].y);
      ua.w = pack_bf16_pair(ra[2*i+1].z, ra[2*i+1].w);
      ub.x = pack_bf16_pair(rb[2*i+0].x, rb[2*i+0].y);
      ub.y = pack_bf16_pair(rb[2*i+0].z, rb[2*i+0].w);
      ub.z = pack_bf16_pair(rb[2*i+1].x, rb[2*i+1].y);
      ub.w = pack_bf16_pair(rb[2*i+1].z, rb[2*i+1].w);
      const int boff = rrow[i] * LDS_PITCH + rc8[i] * 16;
      *(uint4*)(la + boff) = ua;
      *(uint4*)(lb + boff) = ub;
    }
  };

  v8f acc[2][4] = {};
  const int NT = K / BK;

  gload(0);
  lstore(0);
  __syncthreads();

  for (int kt = 0; kt < NT; ++kt) {
    const int cur = kt & 1;
    if (kt + 1 < NT) gload(kt + 1);   // prefetch next tile into registers

    const char* la = lds + cur * STAGE_BYTES;
    const char* lb = la + TILE_BYTES;

    // A fragment (16x32 bf16): lanes 0-15 hold K{0-7,16-23}, lanes 16-31 K{8-15,24-31}
    FragBF af[2], bfr[4];
    #pragma unroll
    for (int mi = 0; mi < 2; ++mi) {
      const int row = wm * 32 + mi * 16 + lmod;
      const char* p = la + row * LDS_PITCH + lhalf * 16;
      af[mi].q[0] = *(const uint4*)(p);
      af[mi].q[1] = *(const uint4*)(p + 32);
    }
    // B fragment (32x16 bf16): lane = column; lanes 0-15 K{0-7,16-23}, lanes 16-31 K{8-15,24-31}
    #pragma unroll
    for (int ni = 0; ni < 4; ++ni) {
      const int row = wn * 64 + ni * 16 + lmod;   // row of W == column of B
      const char* p = lb + row * LDS_PITCH + lhalf * 16;
      bfr[ni].q[0] = *(const uint4*)(p);
      bfr[ni].q[1] = *(const uint4*)(p + 32);
    }

    #pragma unroll
    for (int mi = 0; mi < 2; ++mi)
      #pragma unroll
      for (int ni = 0; ni < 4; ++ni)
        acc[mi][ni] = __builtin_amdgcn_wmma_f32_16x16x32_bf16(
            false, af[mi].v, false, bfr[ni].v, (short)0, acc[mi][ni], false, false);

    if (kt + 1 < NT) lstore(1 - cur);
    __syncthreads();
  }

  // Epilogue: C VGPR j holds M = j (lanes 0-15) or M = 8+j (lanes 16-31), N = lane%16
  const int gm0 = bm * BM + wm * 32;
  const int gn0 = bn * BN + wn * 64;
  #pragma unroll
  for (int mi = 0; mi < 2; ++mi) {
    #pragma unroll
    for (int ni = 0; ni < 4; ++ni) {
      #pragma unroll
      for (int j = 0; j < 8; ++j) {
        const int m = gm0 + mi * 16 + lhalf * 8 + j;
        const int n = gn0 + ni * 16 + lmod;
        float v = acc[mi][ni][j];
        if (MODE == 0) {
          v += bias[n] + pos[(m & 63) * 1024 + n];
        } else if (MODE == 1) {
          v *= kk[((m >> 6) * 8 + (n >> 7)) * 64 + (m & 63)];
        } else {
          v += bias[n];
        }
        Cout[(size_t)m * N + n] = v;
      }
    }
  }
}

// Wqm[h,d] = mean over 128 rows of Wq within head h (folds AvgPool into weight)
__global__ void wqm_kernel(const float* __restrict__ Wq, float* __restrict__ Wqm) {
  const int idx = blockIdx.x * 256 + threadIdx.x;    // 8192 = 8*1024
  const int h = idx >> 10, d = idx & 1023;
  const float* p = Wq + (size_t)h * 128 * 1024 + d;
  float s = 0.f;
  #pragma unroll 8
  for (int j = 0; j < 128; ++j) s += p[(size_t)j * 1024];
  Wqm[idx] = s * (1.0f / 128.0f);
}

// qm[r,h] = xe[r,:] . Wqm[h,:]   (one block per row)
__global__ void qmean_kernel(const float* __restrict__ xe, const float* __restrict__ Wqm,
                             float* __restrict__ qm) {
  __shared__ float red[8 * 256];
  const int r = blockIdx.x, t = threadIdx.x;
  const float* xr = xe + (size_t)r * 1024;
  float acc[8] = {};
  for (int d = t; d < 1024; d += 256) {
    const float x = xr[d];
    #pragma unroll
    for (int h = 0; h < 8; ++h) acc[h] += x * Wqm[h * 1024 + d];
  }
  #pragma unroll
  for (int h = 0; h < 8; ++h) red[h * 256 + t] = acc[h];
  __syncthreads();
  if (t < 8) {
    float s = 0.f;
    for (int i = 0; i < 256; ++i) s += red[t * 256 + i];
    qm[r * 8 + t] = s;
  }
}

// kk[b,h,f] = sigmoid( sum_c qm[b*64+c, h] * Wk[f,c] )
__global__ void kproj_kernel(const float* __restrict__ qm, const float* __restrict__ Wk,
                             float* __restrict__ kk) {
  const int idx = blockIdx.x * 256 + threadIdx.x;    // 8192 = 16*8*64
  const int b = idx >> 9, h = (idx >> 6) & 7, f = idx & 63;
  float s = 0.f;
  #pragma unroll
  for (int c = 0; c < 64; ++c)
    s += qm[((b << 6) + c) * 8 + h] * Wk[f * 64 + c];
  kk[idx] = 1.0f / (1.0f + __expf(-s));
}

extern "C" void kernel_launch(void* const* d_in, const int* in_sizes, int n_in,
                              void* d_out, int out_size, void* d_ws, size_t ws_size,
                              hipStream_t stream) {
  const float* x   = (const float*)d_in[0];   // [16,1,64,128,128] -> [1024,16384]
  const float* We  = (const float*)d_in[1];   // [1024,16384]
  const float* be  = (const float*)d_in[2];   // [1024]
  const float* pos = (const float*)d_in[3];   // [1,64,1024]
  const float* Wq  = (const float*)d_in[4];   // [1024,1024]
  const float* Wk  = (const float*)d_in[5];   // [64,64]
  const float* Wv  = (const float*)d_in[6];   // [1024,1024]
  const float* Wo  = (const float*)d_in[7];   // [16384,1024]
  const float* bo  = (const float*)d_in[8];   // [16384]
  float* out = (float*)d_out;                 // [1024,16384]

  // workspace layout (needs ~8.1 MB)
  char*  ws   = (char*)d_ws;
  float* xe_f = (float*)ws;                     // 1024x1024 fp32  @ 0      (4 MB)
  float* sc_f = (float*)(ws + (4u << 20));      // 1024x1024 fp32  @ 4 MB   (4 MB)
  float* Wqm  = (float*)(ws + (8u << 20));      // 8x1024          (32 KB)
  float* qm   = Wqm + 8192;                     // 1024x8          (32 KB)
  float* kk   = qm + 8192;                      // 16x8x64         (32 KB)

  dim3 blk(256);
  // xe = x @ We^T + be + pos
  gemm_tn_wmma_bf16<0><<<dim3(8, 8), blk, 0, stream>>>(x, We, 16384, 1024, xe_f, be, pos, nullptr);
  // head-gate path (tiny)
  wqm_kernel  <<<32,   256, 0, stream>>>(Wq, Wqm);
  qmean_kernel<<<1024, 256, 0, stream>>>(xe_f, Wqm, qm);
  kproj_kernel<<<32,   256, 0, stream>>>(qm, Wk, kk);
  // scaled = (xe @ Wv^T) * k
  gemm_tn_wmma_bf16<1><<<dim3(8, 8), blk, 0, stream>>>(xe_f, Wv, 1024, 1024, sc_f, nullptr, nullptr, kk);
  // out = scaled @ Wo^T + bo
  gemm_tn_wmma_bf16<2><<<dim3(128, 8), blk, 0, stream>>>(sc_f, Wo, 1024, 16384, out, bo, nullptr, nullptr);
}
